// DeformableConvNet_63333587747326
// MI455X (gfx1250) — matrix operand order repaired and checked by reading, not compiled, not run on hardware
//
#include <hip/hip_runtime.h>
#include <hip/hip_bf16.h>

typedef __attribute__((ext_vector_type(16))) _Float16 v16h;
typedef __attribute__((ext_vector_type(8)))  float    v8f;
typedef __attribute__((ext_vector_type(4)))  float    f4;

#define BB 8
#define CC 64
#define HH 128
#define WW 128
#define OO 64
#define NPIX (BB*HH*WW)   // 131072 pixels; W=128 so every 16-pixel tile stays in one row

// ---------------------------------------------------------------------------
// Kernel 1: NCHW -> NHWC transpose of x (so channel gathers are contiguous)
// ---------------------------------------------------------------------------
__global__ void k_nchw2nhwc(const float* __restrict__ x, float* __restrict__ xt) {
  long i = (long)blockIdx.x * blockDim.x + threadIdx.x;
  if (i >= (long)BB * CC * HH * WW) return;
  int w = i & (WW - 1);
  long t = i >> 7;
  int h = t & (HH - 1); t >>= 7;
  int c = t % CC;
  int b = (int)(t / CC);
  xt[((long)(b * HH + h) * WW + w) * CC + c] = x[i];
}

// ---------------------------------------------------------------------------
// Kernel 2: pack w_dc (O,C,3,3) into WMMA B-fragment order (f16).
// Fragment index t = ((kk*2+cb)*4 + nt)*32 + lane ; element j = K slot j.
// B 32x16 layout: lane 0-15 -> N=lane, K=0..7,16..23 ; lane 16-31 -> N=lane-16, K=8..15,24..31
// ---------------------------------------------------------------------------
__global__ void k_pack_w(const float* __restrict__ w_dc, _Float16* __restrict__ bp) {
  int t = blockIdx.x * blockDim.x + threadIdx.x;
  if (t >= 9 * 2 * 4 * 32) return;
  int lane = t & 31;
  int nt   = (t >> 5) & 3;
  int cb   = (t >> 7) & 1;
  int kk   = t >> 8;
  int N = nt * 16 + (lane & 15);
  int kbase = (lane >> 4) << 3;            // 0 or 8
  #pragma unroll
  for (int j = 0; j < 16; ++j) {
    int c = cb * 32 + kbase + j + ((j >> 3) << 3);
    bp[t * 16 + j] = (_Float16)w_dc[((N * CC) + c) * 9 + kk];
  }
}

// ---------------------------------------------------------------------------
// Kernel 3: offset conv (3x3, C=64 -> 18), weights staged in LDS.
// ---------------------------------------------------------------------------
__global__ void __launch_bounds__(256) k_offconv(
    const float* __restrict__ xt, const float* __restrict__ w_off,
    const float* __restrict__ b_off, float* __restrict__ offs) {
  __shared__ float wl[CC * 9 * 18];                 // wl[(c*9+kk)*18 + o]
  for (int i = threadIdx.x; i < CC * 9 * 18; i += 256) {
    int o = i % 18; int r = i / 18; int kk = r % 9; int c = r / 9;
    wl[i] = w_off[(o * CC + c) * 9 + kk];
  }
  __syncthreads();

  int p = blockIdx.x * blockDim.x + threadIdx.x;    // exactly NPIX threads
  int wp = p & (WW - 1), h = (p >> 7) & (HH - 1), b = p >> 14;

  float acc[18];
  #pragma unroll
  for (int o = 0; o < 18; ++o) acc[o] = b_off[o];

  for (int ki = 0; ki < 3; ++ki) {
    int y = h + ki - 1;
    if ((unsigned)y >= HH) continue;
    for (int kj = 0; kj < 3; ++kj) {
      int xx = wp + kj - 1;
      if ((unsigned)xx >= WW) continue;
      const float* src = xt + ((long)(b * HH + y) * WW + xx) * CC;
      int kk = ki * 3 + kj;
      for (int c = 0; c < CC; ++c) {
        float v = src[c];
        const float* wc_ = &wl[(c * 9 + kk) * 18];
        #pragma unroll
        for (int o = 0; o < 18; ++o) acc[o] = fmaf(v, wc_[o], acc[o]);
      }
    }
  }
  long base = (long)b * 18 * HH * WW + (long)h * WW + wp;
  #pragma unroll
  for (int o = 0; o < 18; ++o) offs[base + (long)o * HH * WW] = acc[o];
}

// ---------------------------------------------------------------------------
// Kernel 4: fused deformable conv.  Each wave owns one 16-pixel M-tile and
// all 4 output-channel N-tiles.  K = tap(9) * 64 channels, consumed in
// 18 chunks of 32 via v_wmma_f32_16x16x32_f16.
// A 16x32 f16 layout: lane L -> M = L&15 ; K slots:
//   lanes 0-15 : j0..7 -> K 0..7 , j8..15 -> K 16..23
//   lanes16-31 : j0..7 -> K 8..15, j8..15 -> K 24..31
// ---------------------------------------------------------------------------
__global__ void __launch_bounds__(256) k_deform(
    const float* __restrict__ xt, const float* __restrict__ offs,
    const _Float16* __restrict__ bp, const float* __restrict__ bias,
    float* __restrict__ out) {
  const int lane = threadIdx.x & 31;
  const int wave = threadIdx.x >> 5;
  const int mt   = blockIdx.x * 8 + wave;           // 8192 M-tiles
  const int half = lane >> 4;
  const int mloc = lane & 15;
  const int pbase = mt * 16;
  const int p  = pbase + mloc;
  const int wp = p & (WW - 1);
  const int h  = (p >> 7) & (HH - 1);
  const int b  = p >> 14;

  const float* op = offs + (long)b * 18 * HH * WW + (long)h * WW + wp;
  const int ostr = HH * WW;

  v8f accs[4] = {};

  for (int kk = 0; kk < 9; ++kk) {
    float dy = op[(2 * kk + 0) * ostr];
    float dx = op[(2 * kk + 1) * ostr];
    float py = (float)h  + (float)(kk / 3 - 1) + dy;
    float px = (float)wp + (float)(kk % 3 - 1) + dx;
    float fy = floorf(py), fx = floorf(px);
    int y0 = (int)fy, x0 = (int)fx;
    float wy = py - fy, wx = px - fx;

    float e[4]; long a[4];
    const int   ys[2]  = { y0, y0 + 1 },  xs[2]  = { x0, x0 + 1 };
    const float wys[2] = { 1.f - wy, wy }, wxs[2] = { 1.f - wx, wx };
    #pragma unroll
    for (int cy = 0; cy < 2; ++cy) {
      #pragma unroll
      for (int cx = 0; cx < 2; ++cx) {
        int yi = ys[cy], xi = xs[cx];
        bool valid = (yi >= 0) & (yi < HH) & (xi >= 0) & (xi < WW);
        int yc = yi < 0 ? 0 : (yi > HH - 1 ? HH - 1 : yi);
        int xc = xi < 0 ? 0 : (xi > WW - 1 ? WW - 1 : xi);
        a[cy * 2 + cx] = ((long)(b * HH + yc) * WW + xc) * CC;
        e[cy * 2 + cx] = valid ? wys[cy] * wxs[cx] : 0.f;
      }
    }

    #pragma unroll
    for (int cb = 0; cb < 2; ++cb) {
      v16h A;
      #pragma unroll
      for (int run = 0; run < 2; ++run) {           // two contiguous 8-channel runs
        int cbase = cb * 32 + half * 8 + run * 16;
        float s[8] = {0, 0, 0, 0, 0, 0, 0, 0};
        #pragma unroll
        for (int cr = 0; cr < 4; ++cr) {            // 4 bilinear corners
          const f4* q = (const f4*)(xt + a[cr] + cbase);
          f4 q0 = q[0], q1 = q[1];
          float ew = e[cr];
          s[0] += q0.x * ew; s[1] += q0.y * ew; s[2] += q0.z * ew; s[3] += q0.w * ew;
          s[4] += q1.x * ew; s[5] += q1.y * ew; s[6] += q1.z * ew; s[7] += q1.w * ew;
        }
        #pragma unroll
        for (int j = 0; j < 8; ++j) A[run * 8 + j] = (_Float16)s[j];
      }

      const v16h* bf = (const v16h*)bp + ((kk * 2 + cb) * 4) * 32 + lane;
      #pragma unroll
      for (int nt = 0; nt < 4; ++nt) {
        v16h Bf = bf[nt * 32];
        accs[nt] = __builtin_amdgcn_wmma_f32_16x16x32_f16(
            false, A, false, Bf, (short)0, accs[nt], false, false);
      }
    }
  }

  // Epilogue: C/D layout -> lane holds N = lane&15, pixels M = r + 8*half.
  #pragma unroll
  for (int nt = 0; nt < 4; ++nt) {
    int och = nt * 16 + mloc;
    float bs = bias[och];
    long obase = ((long)(b * OO + och) * HH + h) * WW;
    int wbase = (pbase & (WW - 1)) + half * 8;
    #pragma unroll
    for (int r = 0; r < 8; ++r)
      out[obase + wbase + r] = accs[nt][r] + bs;
  }
}

// ---------------------------------------------------------------------------
extern "C" void kernel_launch(void* const* d_in, const int* in_sizes, int n_in,
                              void* d_out, int out_size, void* d_ws, size_t ws_size,
                              hipStream_t stream) {
  const float* x     = (const float*)d_in[0];
  const float* w_off = (const float*)d_in[1];
  const float* b_off = (const float*)d_in[2];
  const float* w_dc  = (const float*)d_in[3];
  const float* b_dc  = (const float*)d_in[4];
  float* out = (float*)d_out;

  // Workspace layout (needs ~43 MB):
  float*    xt   = (float*)d_ws;                                // 8.39M floats
  float*    offs = xt + (size_t)BB * HH * WW * CC;              // 2.36M floats
  _Float16* bp   = (_Float16*)(offs + (size_t)BB * 18 * HH * WW); // 36864 f16 (32B-aligned)

  long nelem = (long)BB * CC * HH * WW;
  k_nchw2nhwc<<<(int)((nelem + 255) / 256), 256, 0, stream>>>(x, xt);
  k_pack_w   <<<9, 256, 0, stream>>>(w_dc, bp);
  k_offconv  <<<NPIX / 256, 256, 0, stream>>>(xt, w_off, b_off, offs);
  k_deform   <<<NPIX / 16 / 8, 256, 0, stream>>>(xt, offs, bp, b_dc, out);
}
